// DIEN_14259291423218
// MI455X (gfx1250) — matrix operand order repaired and checked by reading, not compiled, not run on hardware
//
#include <hip/hip_runtime.h>
#include <cstdint>

#define B_ 1024
#define T_ 200

typedef __attribute__((ext_vector_type(16))) _Float16 v16h;
typedef __attribute__((ext_vector_type(8)))  _Float16 v8h;
typedef __attribute__((ext_vector_type(8)))  float    v8f;
typedef __attribute__((ext_vector_type(4)))  unsigned int v4u;
typedef __attribute__((ext_vector_type(8)))  int      v8i;
typedef __attribute__((ext_vector_type(4)))  int      v4i;

__device__ __forceinline__ float sigmoidf_(float x) { return 1.f / (1.f + __expf(-x)); }

__device__ __forceinline__ v8f wmma16x16x32(const v16h& a, const v16h& b, const v8f& c) {
  return __builtin_amdgcn_wmma_f32_16x16x32_f16(false, a, false, b, (short)0, c, false, false);
}

// A fragment (16x32 f16, MxK) from row-major source, leading dim ld, k-chunk base kb.
// Each lane's 16 halves are two contiguous 8-half runs -> two b128 loads.
__device__ __forceinline__ void load_a16(const _Float16* S, int ld, int kb, v16h& a) {
  const int lane = threadIdx.x & 31;
  const _Float16* row = S + (size_t)(lane & 15) * ld + kb + ((lane < 16) ? 0 : 8);
  const v8h lo = *(const v8h*)(row);
  const v8h hi = *(const v8h*)(row + 16);
#pragma unroll
  for (int v = 0; v < 8; ++v) { a[v] = lo[v]; a[8 + v] = hi[v]; }
}

// B fragment from PRE-PACKED weights (fragment-major): two contiguous b128 loads.
// pack index: ((kc*(N/16)+nt)*32 + lane)*16 + v  ==  W[kc*32 + (lane<16?0:8) + (v<8?v:v+8)][nt*16 + (lane&15)]
__device__ __forceinline__ void load_bp(const _Float16* Wp, int N, int kb, int n0, v16h& b) {
  const int lane = threadIdx.x & 31;
  const size_t fi = (((size_t)(kb >> 5) * (N >> 4) + (n0 >> 4)) * 32 + lane) * 16;
  const v8h lo = *(const v8h*)(Wp + fi);
  const v8h hi = *(const v8h*)(Wp + fi + 8);
#pragma unroll
  for (int v = 0; v < 8; ++v) { b[v] = lo[v]; b[8 + v] = hi[v]; }
}

// C/D (16x16 f32): lane n = n0+(lane&15); VGPR v -> row ((lane<16)?0:8)+v.
__device__ __forceinline__ void store_c_f32(float* C, int ldc, int n0, const v8f& c) {
  const int lane = threadIdx.x & 31;
  const int n = n0 + (lane & 15);
  const int mb = (lane < 16) ? 0 : 8;
#pragma unroll
  for (int v = 0; v < 8; ++v) C[(size_t)(mb + v) * ldc + n] = c[v];
}
__device__ __forceinline__ void store_c_f16(_Float16* C, int ldc, int n0, const v8f& c) {
  const int lane = threadIdx.x & 31;
  const int n = n0 + (lane & 15);
  const int mb = (lane < 16) ? 0 : 8;
#pragma unroll
  for (int v = 0; v < 8; ++v) C[(size_t)(mb + v) * ldc + n] = (_Float16)c[v];
}

__device__ __forceinline__ v8f bias_c(const float* bias, int n0) {
  const float bv = bias[n0 + (threadIdx.x & 15)];
  v8f c;
#pragma unroll
  for (int v = 0; v < 8; ++v) c[v] = bv;
  return c;
}
__device__ __forceinline__ v8f zero_c() {
  v8f c;
#pragma unroll
  for (int v = 0; v < 8; ++v) c[v] = 0.f;
  return c;
}
__device__ __forceinline__ void prelu_c(v8f& c) {
#pragma unroll
  for (int v = 0; v < 8; ++v) c[v] = c[v] > 0.f ? c[v] : 0.25f * c[v];
}

// --- Tensor Data Mover: 2D f16 tile (rows x row_elems) global -> LDS, per ISA D# layout. ---
__device__ __forceinline__ void tdm_load_f16(unsigned int lds_addr, const _Float16* gsrc,
                                             unsigned int row_elems, unsigned int rows,
                                             unsigned long long row_stride_elems) {
  unsigned long long ga = (unsigned long long)(uintptr_t)gsrc;
  v4u g0;
  g0[0] = 1u;                                           // count=1, user mode
  g0[1] = lds_addr;                                     // LDS byte address
  g0[2] = (unsigned int)ga;                             // global_addr[31:0]
  g0[3] = (unsigned int)((ga >> 32) & 0x1FFFFFFull) | (2u << 30);  // addr[56:32] | type=2
  const unsigned int td0 = row_elems, td1 = rows, tl0 = row_elems, tl1 = rows;
  const unsigned long long s0 = row_stride_elems, s1 = 0ull;
  v8i g1;
  g1[0] = (int)(1u << 16);                              // data_size=1 (2 bytes)
  g1[1] = (int)((td0 & 0xFFFFu) << 16);                 // tensor_dim0[15:0]
  g1[2] = (int)((td0 >> 16) | ((td1 & 0xFFFFu) << 16)); // td0 hi | td1 lo
  g1[3] = (int)((td1 >> 16) | (tl0 << 16));             // td1 hi | tile_dim0
  g1[4] = (int)(tl1);                                   // tile_dim1 | tile_dim2=0
  g1[5] = (int)(s0 & 0xFFFFFFFFull);                    // dim0 stride lo
  g1[6] = (int)((s0 >> 32) | ((s1 & 0xFFFFull) << 16));
  g1[7] = (int)(s1 >> 16);
  v4i g2 = {0, 0, 0, 0};
  v4i g3 = {0, 0, 0, 0};
  v8i g4 = {0, 0, 0, 0, 0, 0, 0, 0};                    // 6-arg toolchain variant
  __builtin_amdgcn_tensor_load_to_lds(g0, g1, g2, g3, g4, 0);
}

// ---------------- embedding gathers ----------------
__global__ __launch_bounds__(256) void k_gather_seq(const int* __restrict__ seq,
                                                    const float* __restrict__ table,
                                                    _Float16* __restrict__ out) {
  size_t i = (size_t)blockIdx.x * 256 + threadIdx.x;   // over B*T*64
  int row = (int)(i >> 6);
  int j = (int)(i & 63);
  int id = seq[row];
  out[i] = (_Float16)table[(size_t)id * 64 + j];
}

__global__ __launch_bounds__(64) void k_gather_misc(const int* __restrict__ item,
                                                    const float* __restrict__ table,
                                                    const float* __restrict__ dense,
                                                    const int* __restrict__ sparse,
                                                    const float* __restrict__ other_tab,
                                                    float* __restrict__ tgt32,
                                                    _Float16* __restrict__ tgt16,
                                                    float* __restrict__ other) {
  const int b = blockIdx.x, j = threadIdx.x;
  int id = item[b];
  float tv = table[(size_t)id * 64 + j];
  tgt32[b * 64 + j] = tv;
  tgt16[b * 64 + j] = (_Float16)tv;
  if (j < 36) {
    float v;
    if (j < 4) v = dense[b * 4 + j];
    else if (j < 20) { int s = sparse[b * 2 + 0]; v = other_tab[(size_t)s * 16 + (j - 4)]; }
    else { int s = sparse[b * 2 + 1]; v = other_tab[(size_t)(1000 + s) * 16 + (j - 20)]; }
    other[b * 36 + j] = v;
  }
}

// ---------------- weight pack: f32 [rows,N] -> f16 WMMA-B fragment order, K padded ----------
__global__ __launch_bounds__(256) void k_pack_b(_Float16* __restrict__ dst,
                                                const float* __restrict__ src,
                                                int rows, int K, int N) {
  const int idx = blockIdx.x * 256 + threadIdx.x;      // fragment-lane index
  const int total = (K >> 5) * (N >> 4) * 32;
  if (idx >= total) return;
  const int lane = idx & 31;
  const int nt = (idx >> 5) % (N >> 4);
  const int kc = (idx >> 5) / (N >> 4);
  const int n = nt * 16 + (lane & 15);
  const int kbase = kc * 32 + ((lane < 16) ? 0 : 8);
#pragma unroll
  for (int v = 0; v < 16; ++v) {
    const int k = kbase + ((v < 8) ? v : v + 8);
    dst[(size_t)idx * 16 + v] = (k < rows) ? (_Float16)src[(size_t)k * N + n] : (_Float16)0.f;
  }
}

// ---------------- fused (AU)GRU scan: 16 batch rows per block, 4 waves ----------------
// W/U fragments are loop-invariant: hoisted into registers, loaded once from packed global.
__global__ __launch_bounds__(128) void k_gru(const _Float16* __restrict__ X,   // [B,T,64] f16
                                             const _Float16* __restrict__ Wp,  // packed [64,192]
                                             const _Float16* __restrict__ Up,  // packed [64,192]
                                             const float* __restrict__ bias,   // [192]
                                             const int* __restrict__ hist,
                                             const float* __restrict__ att,    // [B,T] or null
                                             _Float16* __restrict__ out_seq,   // [B,T,64] or null
                                             float* __restrict__ out_final) {  // [B,64] or null
  __shared__ __align__(16) _Float16 sX[16 * 64];
  __shared__ __align__(16) _Float16 sHh[16 * 64];
  __shared__ float sH[16 * 64];
  __shared__ float sXW[16 * 192];
  __shared__ float sHU[16 * 192];

  const int tid = threadIdx.x;
  const int wave = tid >> 5;
  const int b0 = blockIdx.x * 16;

  v16h wfrag[3][2], ufrag[3][2];
#pragma unroll
  for (int nt = 0; nt < 3; ++nt)
#pragma unroll
    for (int kc = 0; kc < 2; ++kc) {
      load_bp(Wp, 192, kc * 32, wave * 48 + nt * 16, wfrag[nt][kc]);
      load_bp(Up, 192, kc * 32, wave * 48 + nt * 16, ufrag[nt][kc]);
    }

  for (int i = tid; i < 16 * 64; i += 128) { sH[i] = 0.f; sHh[i] = (_Float16)0.f; }
  __syncthreads();

  const unsigned int sx_lds = (unsigned int)(uintptr_t)&sX[0];

  for (int t = 0; t < T_; ++t) {
    if (tid < 32) {  // wave 0 issues the TDM tile fetch: 16 rows x 64 f16, row stride T*64
      tdm_load_f16(sx_lds, X + ((size_t)b0 * T_ + t) * 64, 64u, 16u,
                   (unsigned long long)T_ * 64);
      __builtin_amdgcn_s_wait_tensorcnt(0);
    }
    __syncthreads();

#pragma unroll
    for (int nt = 0; nt < 3; ++nt) {
      const int n0 = wave * 48 + nt * 16;
      v8f cx = bias_c(bias, n0);  // xw + b
      v8f ch = zero_c();          // hu (kept separate: gate 2 needs r*hu)
#pragma unroll
      for (int kc = 0; kc < 2; ++kc) {
        v16h a;
        load_a16(sX, 64, kc * 32, a);
        cx = wmma16x16x32(a, wfrag[nt][kc], cx);
        load_a16(sHh, 64, kc * 32, a);
        ch = wmma16x16x32(a, ufrag[nt][kc], ch);
      }
      store_c_f32(sXW, 192, n0, cx);
      store_c_f32(sHU, 192, n0, ch);
    }
    __syncthreads();

    for (int e = tid; e < 16 * 64; e += 128) {
      const int r = e >> 6, j = e & 63;
      const int b = b0 + r;
      float xw0 = sXW[r * 192 + j],        hu0 = sHU[r * 192 + j];
      float xw1 = sXW[r * 192 + 64 + j],   hu1 = sHU[r * 192 + 64 + j];
      float xw2 = sXW[r * 192 + 128 + j],  hu2 = sHU[r * 192 + 128 + j];
      float z  = sigmoidf_(xw0 + hu0);
      float rr = sigmoidf_(xw1 + hu1);
      float hh = tanhf(xw2 + rr * hu2);
      if (att) z *= att[(size_t)b * T_ + t];
      float h = sH[e];
      float hn = (1.f - z) * h + z * hh;
      if (t < hist[b]) h = hn;
      sH[e] = h;
      sHh[e] = (_Float16)h;
      if (out_seq) out_seq[((size_t)b * T_ + t) * 64 + j] = (_Float16)h;
    }
    __syncthreads();
  }
  if (out_final)
    for (int e = tid; e < 16 * 64; e += 128)
      out_final[(size_t)(b0 + (e >> 6)) * 64 + (e & 63)] = sH[e];
}

// ---------------- fused attention MLP: feats(256) -> 64 -> 16 -> 1 per 16-row tile ----------
__global__ __launch_bounds__(128) void k_att(const _Float16* __restrict__ tgt16,
                                             const _Float16* __restrict__ seqh16,
                                             const _Float16* __restrict__ W1p,  // packed [256,64]
                                             const float* __restrict__ b1,
                                             const _Float16* __restrict__ W2p,  // packed [64,16]
                                             const float* __restrict__ b2,
                                             const float* __restrict__ W3,      // [16]
                                             const float* __restrict__ b3,
                                             const int* __restrict__ hist,
                                             float* __restrict__ logits) {
  __shared__ __align__(16) _Float16 sA1[16 * 64];
  __shared__ float sA2[16 * 16];
  const int tid = threadIdx.x;
  const int wave = tid >> 5;
  const int lane = tid & 31;
  const int m0 = blockIdx.x * 16;

  const size_t row = (size_t)m0 + (lane & 15);
  const int bb = (int)(row / T_);
  const _Float16* tg = tgt16 + (size_t)bb * 64;
  const _Float16* sh = seqh16 + row * 64;
  const int kh = (lane < 16) ? 0 : 8;

  const int n0 = wave * 16;  // 4 waves cover N=64
  v8f c = bias_c(b1, n0);
#pragma unroll
  for (int kb = 0; kb < 256; kb += 32) {
    // each 32-K chunk lies inside one feature block: sel = qt / h / qt-h / qt*h
    const int sel = kb >> 6;
    const int jb = (kb & 63) + kh;
    const v8h q0 = *(const v8h*)(tg + jb);
    const v8h q1 = *(const v8h*)(tg + jb + 16);
    const v8h s0 = *(const v8h*)(sh + jb);
    const v8h s1 = *(const v8h*)(sh + jb + 16);
    v8h lo, hi;
    if (sel == 0)      { lo = q0;      hi = q1; }
    else if (sel == 1) { lo = s0;      hi = s1; }
    else if (sel == 2) { lo = q0 - s0; hi = q1 - s1; }
    else               { lo = q0 * s0; hi = q1 * s1; }
    v16h a;
#pragma unroll
    for (int v = 0; v < 8; ++v) { a[v] = lo[v]; a[8 + v] = hi[v]; }
    v16h bf;
    load_bp(W1p, 64, kb, n0, bf);
    c = wmma16x16x32(a, bf, c);
  }
  prelu_c(c);
  store_c_f16(sA1, 64, n0, c);
  __syncthreads();

  if (tid < 32) {  // layer 2: [16,64]@[64,16]
    v8f c2 = bias_c(b2, 0);
#pragma unroll
    for (int kb = 0; kb < 64; kb += 32) {
      v16h a, bf;
      load_a16(sA1, 64, kb, a);
      load_bp(W2p, 16, kb, 0, bf);
      c2 = wmma16x16x32(a, bf, c2);
    }
    prelu_c(c2);
    store_c_f32(sA2, 16, 0, c2);
  }
  __syncthreads();

  if (tid < 16) {  // layer 3: 16-wide dot + mask
    const size_t r2 = (size_t)m0 + tid;
    const int b = (int)(r2 / T_);
    const int t = (int)(r2 % T_);
    float acc = b3[0];
#pragma unroll
    for (int i = 0; i < 16; ++i) acc += sA2[tid * 16 + i] * W3[i];
    logits[(size_t)b * T_ + t] = (t < hist[b]) ? acc : -1e9f;
  }
}

__global__ __launch_bounds__(256) void k_softmax(const float* __restrict__ logits,
                                                 float* __restrict__ scores) {
  const int b = blockIdx.x;
  __shared__ float red[256];
  const float* L = logits + (size_t)b * T_;
  float m = -1e30f;
  for (int t = threadIdx.x; t < T_; t += 256) m = fmaxf(m, L[t]);
  red[threadIdx.x] = m; __syncthreads();
  for (int s = 128; s > 0; s >>= 1) {
    if (threadIdx.x < s) red[threadIdx.x] = fmaxf(red[threadIdx.x], red[threadIdx.x + s]);
    __syncthreads();
  }
  m = red[0]; __syncthreads();
  float sum = 0.f;
  for (int t = threadIdx.x; t < T_; t += 256) sum += __expf(L[t] - m);
  red[threadIdx.x] = sum; __syncthreads();
  for (int s = 128; s > 0; s >>= 1) {
    if (threadIdx.x < s) red[threadIdx.x] += red[threadIdx.x + s];
    __syncthreads();
  }
  const float inv = 1.f / red[0];
  for (int t = threadIdx.x; t < T_; t += 256) scores[(size_t)b * T_ + t] = __expf(L[t] - m) * inv;
}

__global__ __launch_bounds__(192) void k_concat(const float* __restrict__ fin,
                                                const float* __restrict__ tgt,
                                                const float* __restrict__ oth,
                                                _Float16* __restrict__ x0) {
  const int b = blockIdx.x, j = threadIdx.x;
  float v;
  if (j < 64) v = fin[b * 64 + j];
  else if (j < 128) v = tgt[b * 64 + (j - 64)];
  else if (j < 164) v = oth[b * 36 + (j - 128)];
  else v = 0.f;                              // K padded to 192
  x0[(size_t)b * 192 + j] = (_Float16)v;
}

// generic WMMA GEMM: C[M,N] = A[M,K] @ Wp[K,N](packed) + b; block=128 (4 waves -> 16x64)
__global__ __launch_bounds__(128) void k_gemm(const _Float16* __restrict__ A, int K,
                                              const _Float16* __restrict__ Wp, int N,
                                              const float* __restrict__ bias,
                                              float* __restrict__ C) {
  const int wave = threadIdx.x >> 5;
  const int m0 = blockIdx.x * 16;
  const int n0 = blockIdx.y * 64 + wave * 16;
  const _Float16* Arow = A + (size_t)m0 * K;
  v8f c = bias_c(bias, n0);
  for (int kb = 0; kb < K; kb += 32) {
    v16h a, bf;
    load_a16(Arow, K, kb, a);
    load_bp(Wp, N, kb, n0, bf);
    c = wmma16x16x32(a, bf, c);
  }
  store_c_f32(C + (size_t)m0 * N, N, n0, c);
}

// DICE: per-column batch mean/var -> p*x ; one block per column
__global__ __launch_bounds__(256) void k_dice(const float* __restrict__ X, int M, int N,
                                              float* __restrict__ Yf, _Float16* __restrict__ Yh) {
  const int col = blockIdx.x;
  __shared__ float s1[256], s2[256];
  float s = 0.f, q = 0.f;
  for (int r = threadIdx.x; r < M; r += 256) {
    float x = X[(size_t)r * N + col]; s += x; q += x * x;
  }
  s1[threadIdx.x] = s; s2[threadIdx.x] = q;
  __syncthreads();
  for (int st = 128; st > 0; st >>= 1) {
    if (threadIdx.x < st) { s1[threadIdx.x] += s1[threadIdx.x + st]; s2[threadIdx.x] += s2[threadIdx.x + st]; }
    __syncthreads();
  }
  const float mean = s1[0] / M;
  const float var = s2[0] / M - mean * mean;
  const float inv = rsqrtf(var + 1e-9f);
  for (int r = threadIdx.x; r < M; r += 256) {
    float x = X[(size_t)r * N + col];
    float y = sigmoidf_((x - mean) * inv) * x;
    if (Yf) Yf[(size_t)r * N + col] = y;
    if (Yh) Yh[(size_t)r * N + col] = (_Float16)y;
  }
}

__global__ __launch_bounds__(256) void k_out(const float* __restrict__ X,
                                             const float* __restrict__ W,
                                             const float* __restrict__ bp,
                                             float* __restrict__ out) {
  const int b = blockIdx.x * 256 + threadIdx.x;
  if (b >= B_) return;
  float acc = bp[0];
#pragma unroll
  for (int j = 0; j < 64; ++j) acc += X[(size_t)b * 64 + j] * W[j];
  out[b] = sigmoidf_(acc);
}

extern "C" void kernel_launch(void* const* d_in, const int* in_sizes, int n_in,
                              void* d_out, int out_size, void* d_ws, size_t ws_size,
                              hipStream_t stream) {
  (void)in_sizes; (void)n_in; (void)out_size; (void)ws_size;
  const float* dense     = (const float*)d_in[0];
  const int*   sparse    = (const int*)d_in[1];
  const int*   seq       = (const int*)d_in[2];
  const int*   item      = (const int*)d_in[3];
  const int*   hist      = (const int*)d_in[4];
  const float* table     = (const float*)d_in[5];
  const float* other_tab = (const float*)d_in[6];
  const float* gruW = (const float*)d_in[7];
  const float* gruU = (const float*)d_in[8];
  const float* gruB = (const float*)d_in[9];
  const float* augW = (const float*)d_in[10];
  const float* augU = (const float*)d_in[11];
  const float* augB = (const float*)d_in[12];
  const float* attW1 = (const float*)d_in[13];
  const float* attB1 = (const float*)d_in[14];
  const float* attW2 = (const float*)d_in[15];
  const float* attB2 = (const float*)d_in[16];
  const float* attW3 = (const float*)d_in[17];
  const float* attB3 = (const float*)d_in[18];
  const float* ffnW1 = (const float*)d_in[19];
  const float* ffnB1 = (const float*)d_in[20];
  const float* ffnW2 = (const float*)d_in[21];
  const float* ffnB2 = (const float*)d_in[22];
  const float* ffnW3 = (const float*)d_in[23];
  const float* ffnB3 = (const float*)d_in[24];
  const float* outW = (const float*)d_in[25];
  const float* outB = (const float*)d_in[26];
  float* out = (float*)d_out;

  char* ws = (char*)d_ws;
  size_t off = 0;
  auto alloc = [&](size_t bytes) { size_t r = off; off += (bytes + 255) & ~(size_t)255; return r; };

  _Float16* gruWp  = (_Float16*)(ws + alloc(64 * 192 * 2));
  _Float16* gruUp  = (_Float16*)(ws + alloc(64 * 192 * 2));
  _Float16* augWp  = (_Float16*)(ws + alloc(64 * 192 * 2));
  _Float16* augUp  = (_Float16*)(ws + alloc(64 * 192 * 2));
  _Float16* attW1p = (_Float16*)(ws + alloc(256 * 64 * 2));
  _Float16* attW2p = (_Float16*)(ws + alloc(64 * 16 * 2));
  _Float16* ffnW1p = (_Float16*)(ws + alloc(192 * 256 * 2));
  _Float16* ffnW2p = (_Float16*)(ws + alloc(256 * 128 * 2));
  _Float16* ffnW3p = (_Float16*)(ws + alloc(128 * 64 * 2));
  _Float16* seqx16 = (_Float16*)(ws + alloc((size_t)B_ * T_ * 64 * 2));
  _Float16* seqh16 = (_Float16*)(ws + alloc((size_t)B_ * T_ * 64 * 2));
  _Float16* tgt16  = (_Float16*)(ws + alloc((size_t)B_ * 64 * 2));
  float*    tgt32  = (float*)(ws + alloc((size_t)B_ * 64 * 4));
  float*    other32= (float*)(ws + alloc((size_t)B_ * 36 * 4));
  float*    logits = (float*)(ws + alloc((size_t)B_ * T_ * 4));
  float*    scores = (float*)(ws + alloc((size_t)B_ * T_ * 4));
  float*    fin32  = (float*)(ws + alloc((size_t)B_ * 64 * 4));
  _Float16* x0_16  = (_Float16*)(ws + alloc((size_t)B_ * 192 * 2));
  float*    h1     = (float*)(ws + alloc((size_t)B_ * 256 * 4));
  _Float16* x1_16  = (_Float16*)(ws + alloc((size_t)B_ * 256 * 2));
  float*    h2     = (float*)(ws + alloc((size_t)B_ * 128 * 4));
  _Float16* x2_16  = (_Float16*)(ws + alloc((size_t)B_ * 128 * 2));
  float*    h3     = (float*)(ws + alloc((size_t)B_ * 64 * 4));

  auto pack = [&](_Float16* dst, const float* src, int rows, int K, int N) {
    int lanes = (K / 32) * (N / 16) * 32;
    k_pack_b<<<(lanes + 255) / 256, 256, 0, stream>>>(dst, src, rows, K, N);
  };
  pack(gruWp, gruW, 64, 64, 192);
  pack(gruUp, gruU, 64, 64, 192);
  pack(augWp, augW, 64, 64, 192);
  pack(augUp, augU, 64, 64, 192);
  pack(attW1p, attW1, 256, 256, 64);
  pack(attW2p, attW2, 64, 64, 16);
  pack(ffnW1p, ffnW1, 164, 192, 256);   // K padded 164 -> 192 with zero rows
  pack(ffnW2p, ffnW2, 256, 256, 128);
  pack(ffnW3p, ffnW3, 128, 128, 64);

  k_gather_seq<<<(size_t)B_ * T_ * 64 / 256, 256, 0, stream>>>(seq, table, seqx16);
  k_gather_misc<<<B_, 64, 0, stream>>>(item, table, dense, sparse, other_tab,
                                       tgt32, tgt16, other32);

  // GRU over sequence, produces seq_h (f16)
  k_gru<<<B_ / 16, 128, 0, stream>>>(seqx16, gruWp, gruUp, gruB, hist,
                                     nullptr, seqh16, nullptr);
  // attention MLP + mask
  k_att<<<B_ * T_ / 16, 128, 0, stream>>>(tgt16, seqh16, attW1p, attB1, attW2p, attB2,
                                          attW3, attB3, hist, logits);
  k_softmax<<<B_, 256, 0, stream>>>(logits, scores);
  // AUGRU over seq_h with attention scores -> final state
  k_gru<<<B_ / 16, 128, 0, stream>>>(seqh16, augWp, augUp, augB, hist,
                                     scores, nullptr, fin32);

  k_concat<<<B_, 192, 0, stream>>>(fin32, tgt32, other32, x0_16);

  k_gemm<<<dim3(B_ / 16, 4), 128, 0, stream>>>(x0_16, 192, ffnW1p, 256, ffnB1, h1);
  k_dice<<<256, 256, 0, stream>>>(h1, B_, 256, nullptr, x1_16);
  k_gemm<<<dim3(B_ / 16, 2), 128, 0, stream>>>(x1_16, 256, ffnW2p, 128, ffnB2, h2);
  k_dice<<<128, 256, 0, stream>>>(h2, B_, 128, nullptr, x2_16);
  k_gemm<<<dim3(B_ / 16, 1), 128, 0, stream>>>(x2_16, 128, ffnW3p, 64, ffnB3, h3);
  k_dice<<<64, 256, 0, stream>>>(h3, B_, 64, h3, nullptr);

  k_out<<<(B_ + 255) / 256, 256, 0, stream>>>(h3, outW, outB, out);
}
